// DeepSeekBlockSparseMoE_49443663512210
// MI455X (gfx1250) — matrix operand order repaired and checked by reading, not compiled, not run on hardware
//
#include <hip/hip_runtime.h>
#include <cstdint>
#include <cstddef>

// Problem constants (match reference).
constexpr int E = 8;      // experts
constexpr int H = 2048;   // hidden
constexpr int F = 1408;   // ffn
constexpr int T = 8192;   // tokens
constexpr int LDG = F + 8; // padded LDS row stride (2832 B -> bank-conflict-free)

// Plain ext-vector types (trivially constructible; safe in unions).
typedef __attribute__((ext_vector_type(16))) __bf16        v16bf;
typedef __attribute__((ext_vector_type(8)))  float         v8f;
typedef __attribute__((ext_vector_type(4)))  unsigned int  u32x4;

__device__ __forceinline__ unsigned short f2bf(float f) {
    unsigned int u = __float_as_uint(f);
    unsigned int r = ((u >> 16) & 1u) + 0x7FFFu;   // round-to-nearest-even
    return (unsigned short)((u + r) >> 16);
}

union FragU { u32x4 q[2]; v16bf v; };

// A-matrix 16x32 bf16 fragment (ISA 7.12.2): lanes 0-15 -> M=lane, K in
// {0..7, 16..23}; lanes 16-31 -> M=lane-16, K in {8..15, 24..31}.
// `row_ptr` already points at this lane's row + k-base.
__device__ __forceinline__ v16bf load_a_frag(const unsigned short* row_ptr, int lane) {
    const unsigned short* q = row_ptr + ((lane & 16) ? 8 : 0);
    FragU f;
    f.q[0] = *reinterpret_cast<const u32x4*>(q);        // K+{0..7}
    f.q[1] = *reinterpret_cast<const u32x4*>(q + 16);   // K+{16..23}
    return f.v;
}

// B-matrix 32x16 bf16 fragment: lanes 0-15 -> N=lane, K in {0..7, 16..23};
// lanes 16-31 -> N=lane-16, K in {8..15, 24..31}. Weights stored [N][K].
__device__ __forceinline__ v16bf load_b_frag(const unsigned short* row_ptr, int lane) {
    const unsigned short* q = row_ptr + ((lane & 16) ? 8 : 0);
    const u32x4* r = reinterpret_cast<const u32x4*>(q);
    FragU f;
    f.q[0] = r[0];
    f.q[1] = r[2];
    return f.v;
}

__device__ __forceinline__ v8f wmma_bf16(v16bf a, v16bf b, v8f c) {
    return __builtin_amdgcn_wmma_f32_16x16x32_bf16(
        /*neg_a=*/false, a, /*neg_b=*/false, b,
        /*c_mod=*/(short)0, c, /*reuse_a=*/false, /*reuse_b=*/false);
}

__device__ __forceinline__ v8f zero8() {
    v8f z = {0.f, 0.f, 0.f, 0.f, 0.f, 0.f, 0.f, 0.f};
    return z;
}

// ---------------------------------------------------------------------------
// fp32 -> bf16 bulk conversion (4 elements / thread).
// ---------------------------------------------------------------------------
__global__ void cvt_bf16_k(const float* __restrict__ in,
                           unsigned short* __restrict__ out, int n4) {
    int i = blockIdx.x * blockDim.x + threadIdx.x;
    if (i >= n4) return;
    float4 v = reinterpret_cast<const float4*>(in)[i];
    ushort4 o;
    o.x = f2bf(v.x); o.y = f2bf(v.y); o.z = f2bf(v.z); o.w = f2bf(v.w);
    reinterpret_cast<ushort4*>(out)[i] = o;
}

// ---------------------------------------------------------------------------
// w2 (E,F,H) fp32 -> bf16 transposed to (E,H,F) so GEMM2's B is [N][K].
// The 32x32 fp32 tile is staged into LDS with CDNA5 async-to-LDS copies
// (global_load_async_to_lds_b128, tracked by ASYNCcnt): one B128 per thread,
// no VGPR round-trip; then read back transposed, convert, and store coalesced.
// ---------------------------------------------------------------------------
__global__ void w2t_k(const float* __restrict__ w2,
                      unsigned short* __restrict__ w2t) {
    __shared__ float tile[32][36];   // 144 B row stride: 16B-aligned B128 LDS writes
    const int e  = blockIdx.z;
    const int h0 = blockIdx.x * 32;
    const int f0 = blockIdx.y * 32;
    const float* src = w2 + (size_t)e * F * H;
    unsigned short* dst = w2t + (size_t)e * H * F;

    const int tid = threadIdx.y * 32 + threadIdx.x;   // 0..255 (32 x 8)
    {
        const int r = tid >> 3;                       // tile row 0..31
        const int c = tid & 7;                        // 16-byte chunk 0..7
        const float* gp = src + (size_t)(f0 + r) * H + h0 + c * 4;
        // Low 32 bits of a generic pointer to LDS == wave-relative LDS offset.
        unsigned lds_off = (unsigned)(uintptr_t)(void*)&tile[r][c * 4];
        asm volatile("global_load_async_to_lds_b128 %0, %1, off"
                     :: "v"(lds_off), "v"(gp)
                     : "memory");
        asm volatile("s_wait_asynccnt 0x0" ::: "memory");
    }
    __syncthreads();

    const int tx = threadIdx.x, ty = threadIdx.y;
    for (int i = ty; i < 32; i += 8)
        dst[(size_t)(h0 + i) * F + f0 + tx] = f2bf(tile[tx][i]);
}

// ---------------------------------------------------------------------------
// Router: logits -> softmax -> top-2 -> renormalize -> per-expert compaction.
// One block (256 threads) per token.
// ---------------------------------------------------------------------------
__global__ void router_k(const float* __restrict__ x,
                         const float* __restrict__ gw,
                         int* __restrict__ counts,
                         int* __restrict__ tok_list,
                         float* __restrict__ tok_wt) {
    const int t   = blockIdx.x;
    const int tid = threadIdx.x;
    float acc[E];
#pragma unroll
    for (int e = 0; e < E; ++e) acc[e] = 0.f;
    const float* xr = x + (size_t)t * H;
    for (int h = tid; h < H; h += 256) {
        float xv = xr[h];
#pragma unroll
        for (int e = 0; e < E; ++e) acc[e] += xv * gw[e * H + h];
    }
    // wave32 butterfly reduce
#pragma unroll
    for (int e = 0; e < E; ++e)
        for (int off = 16; off > 0; off >>= 1)
            acc[e] += __shfl_xor(acc[e], off, 32);
    __shared__ float sred[8][E];
    const int w = tid >> 5, lane = tid & 31;
    if (lane == 0)
#pragma unroll
        for (int e = 0; e < E; ++e) sred[w][e] = acc[e];
    __syncthreads();
    if (tid == 0) {
        float l[E];
#pragma unroll
        for (int e = 0; e < E; ++e) {
            float s = 0.f;
            for (int ww = 0; ww < 8; ++ww) s += sred[ww][e];
            l[e] = s;
        }
        float m = l[0];
#pragma unroll
        for (int e = 1; e < E; ++e) m = fmaxf(m, l[e]);
        float p[E];
#pragma unroll
        for (int e = 0; e < E; ++e) p[e] = __expf(l[e] - m);
        int i0 = 0;
#pragma unroll
        for (int e = 1; e < E; ++e) if (p[e] > p[i0]) i0 = e;
        int i1 = (i0 == 0) ? 1 : 0;
#pragma unroll
        for (int e = 0; e < E; ++e) if (e != i0 && p[e] > p[i1]) i1 = e;
        float s2 = p[i0] + p[i1];
        int s0 = atomicAdd(&counts[i0], 1);
        tok_list[(size_t)i0 * T + s0] = t;
        tok_wt [(size_t)i0 * T + s0] = p[i0] / s2;
        int s1 = atomicAdd(&counts[i1], 1);
        tok_list[(size_t)i1 * T + s1] = t;
        tok_wt [(size_t)i1 * T + s1] = p[i1] / s2;
    }
}

// ---------------------------------------------------------------------------
// Fused expert FFN: per (expert, 64-token tile).
// Pass 1: G = silu(X W1^T) * (X W3^T)  -> bf16 in LDS (64 x F, padded).
// Pass 2: OUT += weight * (G @ W2)      -> f32 atomic scatter.
// 8 waves; each owns a 32x32 macro-tile (2x2 WMMA frags).
// ---------------------------------------------------------------------------
__global__ __launch_bounds__(256, 1) void moe_ffn_k(
    const unsigned short* __restrict__ xb,
    const unsigned short* __restrict__ wb1,
    const unsigned short* __restrict__ wb3,
    const unsigned short* __restrict__ wb2t,
    const int* __restrict__ counts,
    const int* __restrict__ tok_list,
    const float* __restrict__ tok_wt,
    float* __restrict__ out) {
    extern __shared__ unsigned short Gs[];   // 64 * LDG bf16 = 181,248 B (< 320 KB/WGP)

    const int e   = blockIdx.y;
    const int cnt = counts[e];
    const int m0  = blockIdx.x * 64;
    if (m0 >= cnt) return;                   // block-uniform: EXEC stays all-1s

    const int tid  = threadIdx.x;
    const int w    = tid >> 5;
    const int lane = tid & 31;
    const int wm   = w & 1;                  // 2 waves over M (64)
    const int wn   = w >> 1;                 // 4 waves over N (128)
    const int lr   = lane & 15;

    const int*   lst = tok_list + (size_t)e * T;
    const float* wts = tok_wt  + (size_t)e * T;

    const int r0 = m0 + wm * 32 + lr;
    const int r1 = r0 + 16;
    const int tok0 = (r0 < cnt) ? lst[r0] : lst[0];
    const int tok1 = (r1 < cnt) ? lst[r1] : lst[0];
    const unsigned short* a0p = xb + (size_t)tok0 * H;
    const unsigned short* a1p = xb + (size_t)tok1 * H;

    const unsigned short* w1e = wb1  + (size_t)e * F * H;
    const unsigned short* w3e = wb3  + (size_t)e * F * H;
    const unsigned short* w2e = wb2t + (size_t)e * H * F;

    // ---------------- Pass 1 ----------------
    for (int nt = 0; nt < F / 128; ++nt) {
        const int nb = nt * 128 + wn * 32;
        const unsigned short* b1r0 = w1e + (size_t)(nb + lr) * H;
        const unsigned short* b1r1 = w1e + (size_t)(nb + 16 + lr) * H;
        const unsigned short* b3r0 = w3e + (size_t)(nb + lr) * H;
        const unsigned short* b3r1 = w3e + (size_t)(nb + 16 + lr) * H;

        v8f acc1[2][2], acc3[2][2];
#pragma unroll
        for (int i = 0; i < 2; ++i)
#pragma unroll
            for (int j = 0; j < 2; ++j) { acc1[i][j] = zero8(); acc3[i][j] = zero8(); }

        for (int k = 0; k < H; k += 32) {
            __builtin_prefetch(b1r0 + k + 256, 0, 3);   // global_prefetch_b8
            __builtin_prefetch(b3r0 + k + 256, 0, 3);
            v16bf a0  = load_a_frag(a0p + k, lane);
            v16bf a1  = load_a_frag(a1p + k, lane);
            v16bf b10 = load_b_frag(b1r0 + k, lane);
            v16bf b11 = load_b_frag(b1r1 + k, lane);
            v16bf b30 = load_b_frag(b3r0 + k, lane);
            v16bf b31 = load_b_frag(b3r1 + k, lane);
            acc1[0][0] = wmma_bf16(a0, b10, acc1[0][0]);
            acc1[0][1] = wmma_bf16(a0, b11, acc1[0][1]);
            acc1[1][0] = wmma_bf16(a1, b10, acc1[1][0]);
            acc1[1][1] = wmma_bf16(a1, b11, acc1[1][1]);
            acc3[0][0] = wmma_bf16(a0, b30, acc3[0][0]);
            acc3[0][1] = wmma_bf16(a0, b31, acc3[0][1]);
            acc3[1][0] = wmma_bf16(a1, b30, acc3[1][0]);
            acc3[1][1] = wmma_bf16(a1, b31, acc3[1][1]);
        }

        // SwiGLU + bf16 store into LDS G.  C/D layout: VGPR v, lanes 0-15 ->
        // (M=v, N=lane); lanes 16-31 -> (M=v+8, N=lane-16).
#pragma unroll
        for (int i = 0; i < 2; ++i) {
            const int Mb = wm * 32 + i * 16 + ((lane & 16) ? 8 : 0);
#pragma unroll
            for (int j = 0; j < 2; ++j) {
                const int Nc = nb + j * 16 + lr;
#pragma unroll
                for (int v = 0; v < 8; ++v) {
                    float h1 = acc1[i][j][v];
                    float h3 = acc3[i][j][v];
                    float g  = (h1 / (1.f + __expf(-h1))) * h3;   // silu(h1)*h3
                    Gs[(Mb + v) * LDG + Nc] = f2bf(g);
                }
            }
        }
    }
    __syncthreads();

    // ---------------- Pass 2 ----------------
    for (int ht = 0; ht < H / 128; ++ht) {
        const int nb = ht * 128 + wn * 32;
        const unsigned short* b2r0 = w2e + (size_t)(nb + lr) * F;
        const unsigned short* b2r1 = w2e + (size_t)(nb + 16 + lr) * F;
        const unsigned short* ga0  = Gs + (wm * 32 + lr) * LDG;
        const unsigned short* ga1  = Gs + (wm * 32 + 16 + lr) * LDG;

        v8f acc[2][2];
#pragma unroll
        for (int i = 0; i < 2; ++i)
#pragma unroll
            for (int j = 0; j < 2; ++j) acc[i][j] = zero8();

        for (int k = 0; k < F; k += 32) {
            __builtin_prefetch(b2r0 + k + 256, 0, 3);
            v16bf a0 = load_a_frag(ga0 + k, lane);   // ds_load path
            v16bf a1 = load_a_frag(ga1 + k, lane);
            v16bf b0 = load_b_frag(b2r0 + k, lane);
            v16bf b1 = load_b_frag(b2r1 + k, lane);
            acc[0][0] = wmma_bf16(a0, b0, acc[0][0]);
            acc[0][1] = wmma_bf16(a0, b1, acc[0][1]);
            acc[1][0] = wmma_bf16(a1, b0, acc[1][0]);
            acc[1][1] = wmma_bf16(a1, b1, acc[1][1]);
        }

        // Weighted atomic scatter into output (token appears in 2 expert lists).
#pragma unroll
        for (int i = 0; i < 2; ++i) {
            const int Mb = wm * 32 + i * 16 + ((lane & 16) ? 8 : 0);
#pragma unroll
            for (int v = 0; v < 8; ++v) {
                const int slot = m0 + Mb + v;
                if (slot < cnt) {
                    const int   tok = lst[slot];
                    const float wgt = wts[slot];
                    float* orow = out + (size_t)tok * H;
#pragma unroll
                    for (int j = 0; j < 2; ++j) {
                        const int Nc = nb + j * 16 + lr;
                        atomicAdd(orow + Nc, acc[i][j][v] * wgt);  // global_atomic_add_f32
                    }
                }
            }
        }
    }
}

// ---------------------------------------------------------------------------
// Launch. Workspace layout (bytes, 256-aligned):
//   counts   @ 0          (256)
//   tok_list @ 256        (E*T*4   = 262,144)
//   tok_wt   @ 262,400    (E*T*4   = 262,144)
//   xb       @ 524,544    (T*H*2   = 33,554,432)
//   wb1      @ 34,078,976 (E*F*H*2 = 46,137,344)
//   wb3      @ 80,216,320 (E*F*H*2)
//   wb2t     @ 126,353,664(E*F*H*2)   total ~164.5 MiB
// ---------------------------------------------------------------------------
extern "C" void kernel_launch(void* const* d_in, const int* in_sizes, int n_in,
                              void* d_out, int out_size, void* d_ws, size_t ws_size,
                              hipStream_t stream) {
    const float* x  = (const float*)d_in[0];
    const float* gw = (const float*)d_in[1];
    const float* w1 = (const float*)d_in[2];
    const float* w2 = (const float*)d_in[3];
    const float* w3 = (const float*)d_in[4];
    float* out = (float*)d_out;

    char* ws = (char*)d_ws;
    int*            counts   = (int*)(ws + 0);
    int*            tok_list = (int*)(ws + 256);
    float*          tok_wt   = (float*)(ws + 262400);
    unsigned short* xb       = (unsigned short*)(ws + 524544);
    unsigned short* wb1      = (unsigned short*)(ws + 34078976);
    unsigned short* wb3      = (unsigned short*)(ws + 80216320);
    unsigned short* wb2t     = (unsigned short*)(ws + 126353664);

    hipMemsetAsync(counts, 0, 256, stream);
    hipMemsetAsync(out, 0, (size_t)T * H * sizeof(float), stream);

    // fp32 -> bf16 conversions (weights then stay L2-resident: 138 MB < 192 MB).
    {
        int n4 = T * H / 4;
        cvt_bf16_k<<<(n4 + 255) / 256, 256, 0, stream>>>(x, xb, n4);
    }
    {
        int n4 = E * F * H / 4;
        cvt_bf16_k<<<(n4 + 255) / 256, 256, 0, stream>>>(w1, wb1, n4);
        cvt_bf16_k<<<(n4 + 255) / 256, 256, 0, stream>>>(w3, wb3, n4);
    }
    w2t_k<<<dim3(H / 32, F / 32, E), dim3(32, 8), 0, stream>>>(w2, wb2t);

    router_k<<<T, 256, 0, stream>>>(x, gw, counts, tok_list, tok_wt);

    moe_ffn_k<<<dim3(T / 64, E), 256, (size_t)64 * LDG * 2, stream>>>(
        xb, wb1, wb3, wb2t, counts, tok_list, tok_wt, out);
}
